// ChunkedLinearDeltaMemory_3032246911474
// MI455X (gfx1250) — matrix operand order repaired and checked by reading, not compiled, not run on hardware
//
#include <hip/hip_runtime.h>

// ---------------- problem constants ----------------
#define BATCH   8
#define SEQ     4096
#define DIM     512            // D_IN == D_OUT
#define CTOK    64             // chunk length (tokens)
#define NCHUNK  64
#define NBLK    8              // workgroups per batch (split D_OUT rows)
#define ROWS    64             // M rows owned per workgroup
#define NTHREAD 256            // 8 waves of 32
#define MAX_LR   0.2f
#define MIN_DEC  0.5f
#define MAX_NORM 30.0f

// ---------------- LDS layout (dynamic) ----------------
#define SLAB_STRIDE 516                        // f32 M slab row stride (bank spread, 16B aligned)
#define BF_STRIDE   520                        // bf16 row stride (16B aligned)
#define SLAB_F      (ROWS * SLAB_STRIDE)       // 33024 f32 = 132096 B
#define MBF_OFF_B   (SLAB_F * 4)
#define MBF_B       (ROWS * BF_STRIDE * 2)     // 66560 B  (scaled bf16 copy of M)
#define CHA_OFF_B   (MBF_OFF_B + MBF_B)
#define CHA_B       (CTOK * BF_STRIDE * 2)     // 66560 B  (bf16 chunk)
#define TAIL_OFF_B  (CHA_OFF_B + CHA_B)
#define TAIL_F      (512 + 64 * 6 + 8)
#define SMEM_BYTES  (TAIL_OFF_B + TAIL_F * 4)  // ~269 KB (< 320 KB WGP LDS)

typedef __attribute__((ext_vector_type(16))) __bf16 v16bf;
typedef __attribute__((ext_vector_type(8)))  __bf16 v8bf;
typedef __attribute__((ext_vector_type(8)))  float  v8f;
typedef __attribute__((ext_vector_type(2)))  float  v2f;

__device__ __forceinline__ float wave_sum(float v) {
#pragma unroll
  for (int m = 16; m > 0; m >>= 1) v += __shfl_xor(v, m, 32);
  return v;
}
__device__ __forceinline__ float sigmoidf(float x) {
  return 1.0f / (1.0f + __expf(-x));
}

// zero the cross-workgroup sync scratch (gSum[512] f32, gCnt[512] u32)
__global__ void delta_mem_zero_kernel(float* gSum, unsigned* gCnt) {
  int t = threadIdx.x;
  if (t < 512) { gSum[t] = 0.0f; gCnt[t] = 0u; }
}

__global__ __launch_bounds__(NTHREAD, 1)
void delta_mem_scan_kernel(const float* __restrict__ x,
                           const float* __restrict__ Minit,
                           const float* __restrict__ etaW,
                           const float* __restrict__ etaB,
                           const float* __restrict__ alW,
                           const float* __restrict__ alB,
                           const float* __restrict__ gW,
                           const float* __restrict__ gB,
                           float* __restrict__ out,       // [B,S,D] then M_fin [B,D,D]
                           float* __restrict__ gSum,      // [B*NCHUNK]
                           unsigned* __restrict__ gCnt) { // [B*NCHUNK]
  extern __shared__ unsigned char smem[];
  float*  slab  = (float*)smem;                       // [ROWS][SLAB_STRIDE] f32 master M
  __bf16* mbf   = (__bf16*)(smem + MBF_OFF_B);        // [ROWS][BF_STRIDE] bf16 scaled copy
  __bf16* chA   = (__bf16*)(smem + CHA_OFF_B);        // [CTOK][BF_STRIDE] bf16 chunk
  float*  tail  = (float*)(smem + TAIL_OFF_B);
  float*  kmean = tail;            // 512
  float*  vsum  = kmean + 512;     // 64
  float*  lerr  = vsum + 64;       // 64
  float*  rowSq = lerr + 64;       // 64
  float*  rowEt = rowSq + 64;      // 64
  float*  rowAl = rowEt + 64;      // 64
  float*  rowRn = rowAl + 64;      // 64
  float*  scal  = rowRn + 64;      // [0]=scale,[1]=sumsq,[2]=gateAcc,[3]=etaAcc,[4]=alAcc,[5]=cM,[6]=cU

  const int t    = threadIdx.x;
  const int wave = t >> 5;
  const int lane = t & 31;
  const int laneLow = lane & 15;
  const int laneHi  = lane >> 4;
  const int b   = blockIdx.x >> 3;     // batch
  const int blk = blockIdx.x & 7;      // row-slab index within batch
  const int col0 = t, col1 = t + NTHREAD;

  const float eB = etaB[0], aB = alB[0], gbB = gB[0];
  const float ew0 = etaW[col0], ew1 = etaW[col1];
  const float aw0 = alW[col0],  aw1 = alW[col1];
  const float gw0 = gW[col0],   gw1 = gW[col1];

  // ---- init: M slab <- M_init rows [blk*64 .. +63], scale = 1 ----
#pragma unroll 4
  for (int i = 0; i < 128; ++i) {
    int idx = i * NTHREAD + t;
    int r = idx >> 9, c = idx & 511;
    slab[r * SLAB_STRIDE + c] = Minit[((size_t)(blk * ROWS + r)) * DIM + c];
  }
  if (t == 0) scal[0] = 1.0f;
  __syncthreads();

  const size_t MOFF = (size_t)BATCH * SEQ * DIM;

  for (int ch = 0; ch < NCHUNK; ++ch) {
    // ---- phase 0: zero per-chunk accumulators + build scaled bf16 copy of M ----
    if (t < 64) { vsum[t] = 0.0f; rowSq[t] = 0.0f; rowEt[t] = 0.0f; rowAl[t] = 0.0f; }
    if (t == 0) { scal[1] = 0.0f; scal[2] = 0.0f; scal[3] = 0.0f; scal[4] = 0.0f; }
    {
      const float sS = scal[0];   // pending clamp scale (stable since previous phase H)
      const int c2 = t * 2;       // two consecutive columns per thread
#pragma unroll 4
      for (int r = 0; r < ROWS; ++r) {
        v2f m = *(const v2f*)(slab + r * SLAB_STRIDE + c2);
        union { __bf16 h[2]; unsigned u; } p;
        p.h[0] = (__bf16)(sS * m[0]);
        p.h[1] = (__bf16)(sS * m[1]);
        *(unsigned*)(mbf + r * BF_STRIDE + c2) = p.u;   // ds_store_b32
      }
    }
    __syncthreads();

    // ---- phase A: load chunk -> bf16 LDS, per-token reductions ----
    for (int e = 0; e < CTOK; ++e) {
      size_t base = ((size_t)b * SEQ + (size_t)ch * CTOK + e) * DIM;
      float v0 = x[base + col0];
      float v1 = x[base + col1];
      chA[e * BF_STRIDE + col0] = (__bf16)v0;
      chA[e * BF_STRIDE + col1] = (__bf16)v1;
      if (ch + 1 < NCHUNK) {
        __builtin_prefetch(x + base + (size_t)CTOK * DIM + col0, 0, 1); // global_prefetch_b8
      }
      float pSq = wave_sum(v0 * v0 + v1 * v1);
      float pEt = wave_sum(v0 * ew0 + v1 * ew1);
      float pAl = wave_sum(v0 * aw0 + v1 * aw1);
      if (lane == 0) {
        atomicAdd(&rowSq[e], pSq);   // ds_add_f32
        atomicAdd(&rowEt[e], pEt);
        atomicAdd(&rowAl[e], pAl);
      }
    }
    __syncthreads();

    // ---- phase B: per-token rnorm + sigmoid means ----
    if (t < CTOK) {
      float nrm = sqrtf(rowSq[t]);
      rowRn[t] = 1.0f / fmaxf(nrm, 1e-5f);
      atomicAdd(&scal[3], sigmoidf(rowEt[t] + eB));
      atomicAdd(&scal[4], sigmoidf(rowAl[t] + aB));
    }
    __syncthreads();

    // ---- phase C: k_mean (normalized-key mean) + gate logit ----
    {
      float km0 = 0.0f, km1 = 0.0f;
      for (int e = 0; e < CTOK; ++e) {
        size_t base = ((size_t)b * SEQ + (size_t)ch * CTOK + e) * DIM;
        float rn = rowRn[e];
        km0 = fmaf(x[base + col0], rn, km0);
        km1 = fmaf(x[base + col1], rn, km1);
      }
      km0 *= (1.0f / CTOK); km1 *= (1.0f / CTOK);
      kmean[col0] = km0; kmean[col1] = km1;
      float g = wave_sum(km0 * gw0 + km1 * gw1);
      if (lane == 0) atomicAdd(&scal[2], g);
    }
    __syncthreads();

    // ---- phase D: fold per-chunk scalars ----
    if (t == 0) {
      float eta   = MAX_LR * scal[3] * (1.0f / CTOK);
      float alpha = MIN_DEC + (1.0f - MIN_DEC) * scal[4] * (1.0f / CTOK);
      float gate  = sigmoidf(scal[2] + gbB);
      scal[5] = scal[0] * (gate * alpha + (1.0f - gate));  // cM (clamp scale folded)
      scal[6] = gate * eta;                                // cU
    }
    __syncthreads();

    // ---- phase E: outputs = chunk @ (scale*M)^T, pure ds_load + WMMA inner loop ----
    {
      const int cT = wave & 3;       // c-tile 0..3 (chunk rows)
      const int oP = wave >> 2;      // o-tile pair 0..1 (local M rows)
      v8f acc0 = {}; v8f acc1 = {};
      // A: lane = chunk row, 16 contiguous K per lane-half
      const __bf16* aBase  = chA + (size_t)(cT * 16 + laneLow) * BF_STRIDE + laneHi * 16;
      // B: lane = M row (output column), 16 contiguous K per lane-half
      const __bf16* bBase0 = mbf + (size_t)(oP * 32 +      laneLow) * BF_STRIDE + laneHi * 16;
      const __bf16* bBase1 = mbf + (size_t)(oP * 32 + 16 + laneLow) * BF_STRIDE + laneHi * 16;
#pragma unroll 4
      for (int ks = 0; ks < 16; ++ks) {
        const int k0 = ks * 32;
        v8bf alo = *(const v8bf*)(aBase + k0);
        v8bf ahi = *(const v8bf*)(aBase + k0 + 8);
        v8bf b0lo = *(const v8bf*)(bBase0 + k0);
        v8bf b0hi = *(const v8bf*)(bBase0 + k0 + 8);
        v8bf b1lo = *(const v8bf*)(bBase1 + k0);
        v8bf b1hi = *(const v8bf*)(bBase1 + k0 + 8);
        v16bf a, bb0, bb1;
#pragma unroll
        for (int j = 0; j < 8; ++j) {
          a[j] = alo[j];   a[j + 8] = ahi[j];
          bb0[j] = b0lo[j]; bb0[j + 8] = b0hi[j];
          bb1[j] = b1lo[j]; bb1[j + 8] = b1hi[j];
        }
        acc0 = __builtin_amdgcn_wmma_f32_16x16x32_bf16(false, a, false, bb0,
                                                       (short)0, acc0, false, false);
        acc1 = __builtin_amdgcn_wmma_f32_16x16x32_bf16(false, a, false, bb1,
                                                       (short)0, acc1, false, false);
      }
      // store outputs + accumulate v_target partials
      const int oG0 = blk * ROWS + oP * 32 + laneLow;
      const int oG1 = oG0 + 16;
      float s0 = 0.0f, s1 = 0.0f;
#pragma unroll
      for (int i = 0; i < 8; ++i) {
        int crow = cT * 16 + i + laneHi * 8;           // WMMA C/D layout
        size_t orow = ((size_t)b * SEQ + (size_t)ch * CTOK + crow) * DIM;
        out[orow + oG0] = acc0[i];
        out[orow + oG1] = acc1[i];
        s0 += acc0[i];
        s1 += acc1[i];
      }
      atomicAdd(&vsum[oP * 32 + laneLow], s0);
      atomicAdd(&vsum[oP * 32 + 16 + laneLow], s1);
    }
    __syncthreads();

    // ---- phase F: Mk = (scale*M) @ k_mean ; err = v_target - Mk (local rows) ----
#pragma unroll
    for (int r = 0; r < 8; ++r) {
      int o = wave * 8 + r;
      float s = 0.0f;
#pragma unroll
      for (int j = 0; j < 16; ++j) {
        int d = lane + j * 32;
        s = fmaf(slab[o * SLAB_STRIDE + d], kmean[d], s);
      }
      s = wave_sum(s);
      if (lane == 0) lerr[o] = vsum[o] * (1.0f / CTOK) - scal[0] * s;
    }
    __syncthreads();

    // ---- phase G: M <- cM*M + cU * err (x) k_mean ; local sumsq ----
    {
      const float cM = scal[5], cU = scal[6];
      float sq = 0.0f;
#pragma unroll 4
      for (int i = 0; i < 128; ++i) {
        int idx = i * NTHREAD + t;
        int r = idx >> 9, c = idx & 511;
        float m  = slab[r * SLAB_STRIDE + c];
        float nm = fmaf(cM, m, cU * lerr[r] * kmean[c]);
        slab[r * SLAB_STRIDE + c] = nm;
        sq = fmaf(nm, nm, sq);
      }
      sq = wave_sum(sq);
      if (lane == 0) atomicAdd(&scal[1], sq);
    }
    __syncthreads();

    // ---- phase H: cross-workgroup Frobenius reduction -> new pending scale ----
    if (t == 0) {
      const int gi = b * NCHUNK + ch;
      __hip_atomic_fetch_add(&gSum[gi], scal[1], __ATOMIC_RELAXED, __HIP_MEMORY_SCOPE_AGENT);
      __hip_atomic_fetch_add(&gCnt[gi], 1u, __ATOMIC_RELEASE, __HIP_MEMORY_SCOPE_AGENT);
      while (__hip_atomic_load(&gCnt[gi], __ATOMIC_ACQUIRE, __HIP_MEMORY_SCOPE_AGENT)
             < (unsigned)NBLK) {
        __builtin_amdgcn_s_sleep(1);
      }
      float tot = __hip_atomic_load(&gSum[gi], __ATOMIC_RELAXED, __HIP_MEMORY_SCOPE_AGENT);
      float nrm = sqrtf(tot);
      scal[0] = fminf(MAX_NORM / (nrm + 1e-6f), 1.0f);
    }
    __syncthreads();
  }

  // ---- final: M_fin = scale * M slab ----
  {
    const float sS = scal[0];
#pragma unroll 4
    for (int i = 0; i < 128; ++i) {
      int idx = i * NTHREAD + t;
      int r = idx >> 9, c = idx & 511;
      out[MOFF + ((size_t)b * DIM + blk * ROWS + r) * DIM + c] =
          sS * slab[r * SLAB_STRIDE + c];
    }
  }
}

extern "C" void kernel_launch(void* const* d_in, const int* in_sizes, int n_in,
                              void* d_out, int out_size, void* d_ws, size_t ws_size,
                              hipStream_t stream) {
  (void)in_sizes; (void)n_in; (void)out_size; (void)ws_size;
  const float* x     = (const float*)d_in[0];
  const float* Minit = (const float*)d_in[1];
  const float* etaW  = (const float*)d_in[2];
  const float* etaB  = (const float*)d_in[3];
  const float* alW   = (const float*)d_in[4];
  const float* alB   = (const float*)d_in[5];
  const float* gW    = (const float*)d_in[6];
  const float* gB    = (const float*)d_in[7];
  float*    out  = (float*)d_out;
  float*    gSum = (float*)d_ws;
  unsigned* gCnt = (unsigned*)d_ws + 512;

  delta_mem_zero_kernel<<<1, 1024, 0, stream>>>(gSum, gCnt);
  delta_mem_scan_kernel<<<dim3(BATCH * NBLK), dim3(NTHREAD), SMEM_BYTES, stream>>>(
      x, Minit, etaW, etaB, alW, alB, gW, gB, out, gSum, gCnt);
}
